// AttentionInteractionNetwork_42314017800754
// MI455X (gfx1250) — matrix operand order repaired and checked by reading, not compiled, not run on hardware
//
#include <hip/hip_runtime.h>

#define NN 10000
#define NE 200000
#define DD 256
#define HH 512

#define MT 128              // rows per block tile
#define NTHREADS 512        // 16 wave32
#define NWAVES 16

typedef __attribute__((ext_vector_type(8)))  float  v8f;
typedef __attribute__((ext_vector_type(8)))  __bf16 v8bf;
typedef __attribute__((ext_vector_type(16))) __bf16 v16bf;

__device__ __forceinline__ v16bf cat16(v8bf a, v8bf b) {
  return __builtin_shufflevector(a, b, 0,1,2,3,4,5,6,7,8,9,10,11,12,13,14,15);
}

// shifted softplus: log(1+exp(x)) - ln2, numerically stable
__device__ __forceinline__ float ssp_f(float x) {
  return fmaxf(x, 0.f) + log1pf(expf(-fabsf(x))) - 0.69314718056f;
}

// ---------------------------------------------------------------------------
// Prep: convert all 6 weight matrices fp32 -> bf16 into workspace
// ---------------------------------------------------------------------------
__global__ void cvt_weights_kernel(const float* __restrict__ e0, const float* __restrict__ e1,
                                   const float* __restrict__ e2, const float* __restrict__ n0,
                                   const float* __restrict__ n1, const float* __restrict__ n2,
                                   __bf16* __restrict__ dst) {
  const int S0 = HH * 3 * DD, S1 = HH * HH, S2 = DD * HH;
  const int TOT = 2 * (S0 + S1 + S2);
  int i = blockIdx.x * blockDim.x + threadIdx.x;
  if (i >= TOT) return;
  int off = i; const float* src;
  if (off < S0)              src = e0 + off;
  else if ((off -= S0) < S1) src = e1 + off;
  else if ((off -= S1) < S2) src = e2 + off;
  else if ((off -= S2) < S0) src = n0 + off;
  else if ((off -= S0) < S1) src = n1 + off;
  else { off -= S1;          src = n2 + off; }
  dst[i] = (__bf16)(*src);
}

__global__ void zero_kernel(float4* __restrict__ p, int n4) {
  int i = blockIdx.x * blockDim.x + threadIdx.x;
  if (i < n4) p[i] = make_float4(0.f, 0.f, 0.f, 0.f);
}

// ---------------------------------------------------------------------------
// Per-edge attention gates: sigmoid(edges . w + b) * cutoff  (one wave/edge)
// ---------------------------------------------------------------------------
__global__ void attn_kernel(const float* __restrict__ edges, const float* __restrict__ cutoff,
                            const float* __restrict__ rw, const float* __restrict__ rb,
                            const float* __restrict__ sw, const float* __restrict__ sb,
                            float* __restrict__ rattn, float* __restrict__ sattn) {
  const int wave = threadIdx.x >> 5, lane = threadIdx.x & 31;
  const int e = blockIdx.x * 8 + wave;
  if (e >= NE) return;
  const float* er = edges + (size_t)e * DD;
  float dr = 0.f, ds = 0.f;
#pragma unroll
  for (int j = 0; j < 8; ++j) {
    const float x = er[lane + 32 * j];
    dr += x * rw[lane + 32 * j];
    ds += x * sw[lane + 32 * j];
  }
#pragma unroll
  for (int off = 16; off > 0; off >>= 1) {
    dr += __shfl_xor(dr, off, 32);
    ds += __shfl_xor(ds, off, 32);
  }
  if (lane == 0) {
    const float c = cutoff[e];
    rattn[e] = c / (1.f + expf(-(dr + rb[0])));
    sattn[e] = c / (1.f + expf(-(ds + sb[0])));
  }
}

// ---------------------------------------------------------------------------
// One GEMM layer of the fused MLP: D(128xNOUT) = act(A(128xK) * W^T + b)
// A in LDS (bf16, row-major, stride K). W global bf16 [NOUT][K] row-major.
// Each wave owns N-column groups (nt strided by 16 waves) and all 8 M-subtiles,
// sharing one B fragment across 8 independent accumulators (deep WMMA chain
// to bury WAR-hazard nops and ds_load latency). Weights read once per block.
// ---------------------------------------------------------------------------
template <int K, int NOUT, bool ACT, bool OUT_F32>
__device__ __forceinline__ void mlp_layer(const __bf16* __restrict__ A,
                                          const __bf16* __restrict__ W,
                                          const float* __restrict__ bias,
                                          __bf16* __restrict__ outb,
                                          float* __restrict__ outf,
                                          int wave, int lane) {
  const int lm = lane & 15;
  const int hi = lane >> 4;
  const __bf16* Ab = A + lm * K + hi * 8;  // A frag: K chunks [kh..kh+7],[kh+16..kh+23]
  for (int nt = wave; nt < NOUT / 16; nt += NWAVES) {
    const int n = nt * 16 + lm;
    const __bf16* Wp = W + (size_t)n * K + hi * 8;  // B frag: K chunks [kh..kh+7],[kh+16..kh+23]
    v8f acc[8];
#pragma unroll
    for (int mt = 0; mt < 8; ++mt) acc[mt] = v8f{};
    for (int k = 0; k < K; k += 32) {
      const v16bf bv = cat16(*(const v8bf*)(Wp + k), *(const v8bf*)(Wp + k + 16));
#pragma unroll
      for (int mt = 0; mt < 8; ++mt) {
        const v16bf av = cat16(*(const v8bf*)(Ab + mt * 16 * K + k),
                               *(const v8bf*)(Ab + mt * 16 * K + k + 16));
        acc[mt] = __builtin_amdgcn_wmma_f32_16x16x32_bf16(false, av, false, bv,
                                                          (short)0, acc[mt], false, false);
      }
    }
    const float bn = bias[n];
#pragma unroll
    for (int mt = 0; mt < 8; ++mt) {
#pragma unroll
      for (int r = 0; r < 8; ++r) {
        float x = acc[mt][r] + bn;
        if (ACT) x = ssp_f(x);
        const int row = mt * 16 + hi * 8 + r;
        if (OUT_F32) outf[row * NOUT + n] = x;
        else         outb[row * NOUT + n] = (__bf16)x;
      }
    }
  }
}

// ---------------------------------------------------------------------------
// Fused 3-layer MLP + LayerNorm + residual (+ attention-weighted scatter-add
// for the edge pass). 512 threads = 16 wave32, 128-row tile, 320KB dynamic LDS
// (full WGP LDS, 16 waves/WGP -- same occupancy as 2x160KB, half the weight
// traffic per row).
// ---------------------------------------------------------------------------
template <bool IS_EDGE>
__global__ __launch_bounds__(NTHREADS)
void mlp_fused_kernel(const float* __restrict__ xin,       // edges[E,256] or nodes[N,256]
                      const float* __restrict__ seg1,      // edge: nodes (senders) | node: recv_agg
                      const float* __restrict__ seg2,      // edge: nodes (receivers) | node: sent_agg
                      const int* __restrict__ senders,
                      const int* __restrict__ receivers,
                      const __bf16* __restrict__ w0, const float* __restrict__ b0,
                      const __bf16* __restrict__ w1, const float* __restrict__ b1,
                      const __bf16* __restrict__ w2, const float* __restrict__ b2,
                      const float* __restrict__ gamma, const float* __restrict__ beta,
                      const float* __restrict__ rattn, const float* __restrict__ sattn,
                      float* __restrict__ sent_agg, float* __restrict__ recv_agg,
                      float* __restrict__ out, int Mtotal) {
  extern __shared__ char smem[];
  __bf16* Xs = (__bf16*)smem;                    // [128][768] bf16 (192 KB); reused [128][512] layer-2 out
  __bf16* Hs = (__bf16*)(smem + MT * 768 * 2);   // [128][512] bf16 (128 KB)
  float*  Us = (float*)(smem + MT * 768 * 2);    // [128][256] f32  overlaps Hs

  const int tid = threadIdx.x;
  const int wave = tid >> 5, lane = tid & 31;
  const int row_base = blockIdx.x * MT;

  // --- assemble concat(x, seg1, seg2) tile into LDS as bf16 (vectorized) ---
  for (int i = tid; i < MT * 192; i += NTHREADS) {
    const int m = i / 192;
    const int c4 = (i - m * 192) * 4;
    const int gm = row_base + m;
    __bf16* dst = Xs + m * 768 + c4;
    if (gm < Mtotal) {
      const float* src;
      if (IS_EDGE) {
        if (c4 < 256)      src = xin  + (size_t)gm * 256 + c4;
        else if (c4 < 512) src = seg1 + (size_t)senders[gm] * 256 + (c4 - 256);
        else               src = seg2 + (size_t)receivers[gm] * 256 + (c4 - 512);
      } else {
        if (c4 < 256)      src = xin  + (size_t)gm * 256 + c4;
        else if (c4 < 512) src = seg1 + (size_t)gm * 256 + (c4 - 256);
        else               src = seg2 + (size_t)gm * 256 + (c4 - 512);
      }
      const float4 v = *(const float4*)src;
      dst[0] = (__bf16)v.x; dst[1] = (__bf16)v.y; dst[2] = (__bf16)v.z; dst[3] = (__bf16)v.w;
    } else {
      dst[0] = (__bf16)0.f; dst[1] = (__bf16)0.f; dst[2] = (__bf16)0.f; dst[3] = (__bf16)0.f;
    }
  }
  __syncthreads();

  mlp_layer<768, 512, true, false>(Xs, w0, b0, Hs, nullptr, wave, lane);
  __syncthreads();
  mlp_layer<512, 512, true, false>(Hs, w1, b1, Xs, nullptr, wave, lane);   // out -> Xs[128][512]
  __syncthreads();
  mlp_layer<512, 256, false, true>(Xs, w2, b2, nullptr, Us, wave, lane);   // out -> Us[128][256] f32
  __syncthreads();

  // --- LayerNorm + residual + (edge) gated scatter-add; one wave per 8 rows ---
  for (int rr = 0; rr < 8; ++rr) {
    const int r = wave * 8 + rr;
    const float* urow = Us + r * 256;
    float vals[8], s = 0.f, s2 = 0.f;
#pragma unroll
    for (int j = 0; j < 8; ++j) {
      const float x = urow[lane + 32 * j];
      vals[j] = x; s += x; s2 += x * x;
    }
#pragma unroll
    for (int off = 16; off > 0; off >>= 1) {
      s  += __shfl_xor(s, off, 32);
      s2 += __shfl_xor(s2, off, 32);
    }
    const float mean = s * (1.f / 256.f);
    const float var  = s2 * (1.f / 256.f) - mean * mean;
    const float rstd = rsqrtf(var + 1e-5f);
    const int gm = row_base + r;
    if (gm < Mtotal) {
      float sa = 0.f, ra = 0.f; int sidx = 0, ridx = 0;
      if (IS_EDGE) {
        sa = sattn[gm]; ra = rattn[gm];
        sidx = senders[gm]; ridx = receivers[gm];
      }
#pragma unroll
      for (int j = 0; j < 8; ++j) {
        const int c = lane + 32 * j;
        const float val = (vals[j] - mean) * rstd * gamma[c] + beta[c];
        out[(size_t)gm * 256 + c] = xin[(size_t)gm * 256 + c] + val;
        if (IS_EDGE) {
          unsafeAtomicAdd(sent_agg + (size_t)sidx * 256 + c, val * sa);
          unsafeAtomicAdd(recv_agg + (size_t)ridx * 256 + c, val * ra);
        }
      }
    }
  }
}

// ---------------------------------------------------------------------------
extern "C" void kernel_launch(void* const* d_in, const int* in_sizes, int n_in,
                              void* d_out, int out_size, void* d_ws, size_t ws_size,
                              hipStream_t stream) {
  (void)in_sizes; (void)n_in; (void)out_size; (void)ws_size;
  const float* nodes  = (const float*)d_in[0];
  const float* edges  = (const float*)d_in[1];
  const float* cutoff = (const float*)d_in[2];
  const float* ew0 = (const float*)d_in[3];
  const float* eb0 = (const float*)d_in[4];
  const float* ew1 = (const float*)d_in[5];
  const float* eb1 = (const float*)d_in[6];
  const float* ew2 = (const float*)d_in[7];
  const float* eb2 = (const float*)d_in[8];
  const float* eg  = (const float*)d_in[9];
  const float* ebt = (const float*)d_in[10];
  const float* nw0 = (const float*)d_in[11];
  const float* nb0 = (const float*)d_in[12];
  const float* nw1 = (const float*)d_in[13];
  const float* nb1 = (const float*)d_in[14];
  const float* nw2 = (const float*)d_in[15];
  const float* nb2 = (const float*)d_in[16];
  const float* ng  = (const float*)d_in[17];
  const float* nbt = (const float*)d_in[18];
  const float* rw  = (const float*)d_in[19];
  const float* rb  = (const float*)d_in[20];
  const float* sw  = (const float*)d_in[21];
  const float* sb  = (const float*)d_in[22];
  const int* senders   = (const int*)d_in[23];
  const int* receivers = (const int*)d_in[24];

  // workspace layout
  char* ws = (char*)d_ws;
  const size_t S0 = (size_t)HH * 3 * DD, S1 = (size_t)HH * HH, S2 = (size_t)DD * HH;
  __bf16* ew0b = (__bf16*)ws;
  __bf16* ew1b = ew0b + S0;
  __bf16* ew2b = ew1b + S1;
  __bf16* nw0b = ew2b + S2;
  __bf16* nw1b = nw0b + S0;
  __bf16* nw2b = nw1b + S1;
  const size_t woff = 2 * (S0 + S1 + S2) * sizeof(__bf16);
  float* rattn    = (float*)(ws + woff);
  float* sattn    = rattn + NE;
  float* sent_agg = sattn + NE;                 // contiguous with recv_agg for one zero pass
  float* recv_agg = sent_agg + (size_t)NN * DD;

  float* nodes_out = (float*)d_out;
  float* edges_out = nodes_out + (size_t)NN * DD;

  const int LDS_BYTES = MT * 768 * 2 + MT * 512 * 2;  // 320 KB (full WGP LDS)
  hipFuncSetAttribute(reinterpret_cast<const void*>(mlp_fused_kernel<true>),
                      hipFuncAttributeMaxDynamicSharedMemorySize, LDS_BYTES);
  hipFuncSetAttribute(reinterpret_cast<const void*>(mlp_fused_kernel<false>),
                      hipFuncAttributeMaxDynamicSharedMemorySize, LDS_BYTES);

  const int TOTW = (int)(2 * (S0 + S1 + S2));
  cvt_weights_kernel<<<(TOTW + 255) / 256, 256, 0, stream>>>(ew0, ew1, ew2, nw0, nw1, nw2, ew0b);
  zero_kernel<<<(2 * NN * DD / 4 + 255) / 256, 256, 0, stream>>>((float4*)sent_agg, 2 * NN * DD / 4);
  attn_kernel<<<NE / 8, 256, 0, stream>>>(edges, cutoff, rw, rb, sw, sb, rattn, sattn);

  mlp_fused_kernel<true><<<(NE + MT - 1) / MT, NTHREADS, LDS_BYTES, stream>>>(
      edges, nodes, nodes, senders, receivers,
      ew0b, eb0, ew1b, eb1, ew2b, eb2, eg, ebt,
      rattn, sattn, sent_agg, recv_agg, edges_out, NE);

  mlp_fused_kernel<false><<<(NN + MT - 1) / MT, NTHREADS, LDS_BYTES, stream>>>(
      nodes, recv_agg, sent_agg, senders, receivers,
      nw0b, nb0, nw1b, nb1, nw2b, nb2, ng, nbt,
      nullptr, nullptr, nullptr, nullptr, nodes_out, NN);
}
